// TernaryLinear_60679297958512
// MI455X (gfx1250) — compile-verified, matrix operand order checked by reading, not run on hardware
//
#include <hip/hip_runtime.h>

// ---------------------------------------------------------------------------
// TernaryLinear for MI455X (gfx1250, wave32, WMMA).
// Y[M,N] = X[M,K] @ Wq[N,K]^T, Wq = ternary(W) in {-1,0,+1} * factor({0,1}).
// X is split into bf16 hi (exact truncation) + bf16 residual; two
// v_wmma_f32_16x16x32_bf16 per (tile, k-step) accumulate into one f32
// accumulator: near-f32 accuracy at ~4x the per-instruction throughput of
// V_WMMA_F32_16X16X4_F32 (16384 vs 2048 FLOP/instr, two passes).
// ---------------------------------------------------------------------------

typedef __bf16 bf16_t;
typedef __bf16 v16bf __attribute__((ext_vector_type(16)));
typedef __bf16 v8bf  __attribute__((ext_vector_type(8)));
typedef float  v8f   __attribute__((ext_vector_type(8)));
typedef float  v4f   __attribute__((ext_vector_type(4)));

#define K_DIM       4096
#define N_DIM       4096
#define BM          128
#define BN          128
#define BK          32
#define LDS_STRIDE  40        // bf16 elems per row: 32 + 8 pad (80B) -> conflict-free b128 gathers
#define NTHREADS    256       // 8 waves of 32
#define RED_BLOCKS  1024

// ---- bf16 helpers (bit ops only; __bf16 used purely as storage) -----------
__device__ __forceinline__ bf16_t f32_to_bf16_rne(float f) {
    unsigned u = __builtin_bit_cast(unsigned, f);
    unsigned r = u + 0x7FFFu + ((u >> 16) & 1u);     // round-to-nearest-even
    return __builtin_bit_cast(bf16_t, (unsigned short)(r >> 16));
}
__device__ __forceinline__ bf16_t f32_hi_bf16(float f) {  // exact truncation
    unsigned u = __builtin_bit_cast(unsigned, f);
    return __builtin_bit_cast(bf16_t, (unsigned short)(u >> 16));
}
__device__ __forceinline__ float f32_trunc16(float f) {   // f with low mantissa zeroed
    unsigned u = __builtin_bit_cast(unsigned, f) & 0xFFFF0000u;
    return __builtin_bit_cast(float, u);
}

struct HiLo { v8bf hi, lo; };

// Split 8 f32 into exact-bf16 hi + RNE-bf16 residual.
__device__ __forceinline__ HiLo split8(v4f a, v4f b) {
    HiLo r;
#pragma unroll
    for (int e = 0; e < 4; ++e) {
        float f0 = a[e], f1 = b[e];
        r.hi[e]     = f32_hi_bf16(f0);
        r.hi[4 + e] = f32_hi_bf16(f1);
        r.lo[e]     = f32_to_bf16_rne(f0 - f32_trunc16(f0));
        r.lo[4 + e] = f32_to_bf16_rne(f1 - f32_trunc16(f1));
    }
    return r;
}

// Ternary-quantize 8 f32 weights into bf16 {-factor, 0, +factor}.
__device__ __forceinline__ v8bf quant8(v4f a, v4f b, bf16_t qp, bf16_t qn, bf16_t qz) {
    v8bf r;
#pragma unroll
    for (int e = 0; e < 4; ++e) {
        float f0 = a[e], f1 = b[e];
        r[e]     = (f0 > 0.5f) ? qp : ((f0 < -0.5f) ? qn : qz);
        r[4 + e] = (f1 > 0.5f) ? qp : ((f1 < -0.5f) ? qn : qz);
    }
    return r;
}

// ---------------------------------------------------------------------------
// Pass 1: per-block partial sums of |W| (deterministic fixed tree).
// ---------------------------------------------------------------------------
__global__ __launch_bounds__(NTHREADS)
void absmean_partial(const float* __restrict__ w, float* __restrict__ partial, int n4) {
    const v4f* w4 = (const v4f*)w;
    float s = 0.0f;
    int stride = gridDim.x * blockDim.x;
    for (int i = blockIdx.x * blockDim.x + threadIdx.x; i < n4; i += stride) {
        v4f v = w4[i];
        s += fabsf(v[0]) + fabsf(v[1]) + fabsf(v[2]) + fabsf(v[3]);
    }
    __shared__ float red[NTHREADS];
    red[threadIdx.x] = s;
    __syncthreads();
    for (int off = NTHREADS / 2; off > 0; off >>= 1) {
        if (threadIdx.x < (unsigned)off) red[threadIdx.x] += red[threadIdx.x + off];
        __syncthreads();
    }
    if (threadIdx.x == 0) partial[blockIdx.x] = red[0];
}

// ---------------------------------------------------------------------------
// Pass 2: reduce partials -> factor = clip(rint(1/(mean+eps)), -1, 1) in {0,1}
// ---------------------------------------------------------------------------
__global__ __launch_bounds__(NTHREADS)
void finalize_factor(const float* __restrict__ partial, float* __restrict__ factor,
                     int nPartial, float invCount) {
    __shared__ float red[NTHREADS];
    float s = 0.0f;
    for (int i = threadIdx.x; i < nPartial; i += blockDim.x) s += partial[i];
    red[threadIdx.x] = s;
    __syncthreads();
    for (int off = NTHREADS / 2; off > 0; off >>= 1) {
        if (threadIdx.x < (unsigned)off) red[threadIdx.x] += red[threadIdx.x + off];
        __syncthreads();
    }
    if (threadIdx.x == 0) {
        float mean = red[0] * invCount;
        float g = 1.0f / (mean + 1e-7f);
        float r = rintf(g);                      // half-to-even, matches jnp.round
        *factor = fminf(fmaxf(r, -1.0f), 1.0f);  // in {0,1} since g > 0
    }
}

// ---------------------------------------------------------------------------
// Pass 3: tiled WMMA GEMM.  Block tile 128x128, BK=32, 8 waves (2M x 4N),
// each wave: 4x2 = 8 accumulator tiles, 16 WMMAs per k-step (hi + lo).
// ---------------------------------------------------------------------------
__global__ __launch_bounds__(NTHREADS)
void ternary_gemm(const float* __restrict__ X, const float* __restrict__ W,
                  const float* __restrict__ factorPtr, float* __restrict__ Y) {
    __shared__ bf16_t sAhi[2][BM * LDS_STRIDE];
    __shared__ bf16_t sAlo[2][BM * LDS_STRIDE];
    __shared__ bf16_t sB  [2][BN * LDS_STRIDE];

    const int tid   = threadIdx.x;
    const int lane  = tid & 31;
    const int wave  = tid >> 5;
    const int waveM = (wave >> 2) * 64;      // 0 or 64
    const int waveN = (wave & 3)  * 32;      // 0,32,64,96
    const int m0    = blockIdx.y * BM;
    const int n0    = blockIdx.x * BN;

    const float factor = *factorPtr;         // uniform scalar load
    const bf16_t qPos  = f32_to_bf16_rne(factor);
    const bf16_t qNeg  = f32_to_bf16_rne(-factor);
    const bf16_t qZero = __builtin_bit_cast(bf16_t, (unsigned short)0);

    // stage-in: each thread handles one 16-float row segment of X and W tiles
    const int row = tid >> 1;                // 0..127
    const int col = (tid & 1) * 16;          // 0 or 16
    const float* xg = X + (size_t)(m0 + row) * K_DIM + col;
    const float* wg = W + (size_t)(n0 + row) * K_DIM + col;

    // Staging kept strictly in named vector registers (no spillable arrays).
    v4f xv0, xv1, xv2, xv3, wv0, wv1, wv2, wv3;

    auto fetch = [&](int kt) {
        const float* xp = xg + kt * BK;
        const float* wp = wg + kt * BK;
        xv0 = *(const v4f*)(xp + 0);  xv1 = *(const v4f*)(xp + 4);
        xv2 = *(const v4f*)(xp + 8);  xv3 = *(const v4f*)(xp + 12);
        wv0 = *(const v4f*)(wp + 0);  wv1 = *(const v4f*)(wp + 4);
        wv2 = *(const v4f*)(wp + 8);  wv3 = *(const v4f*)(wp + 12);
    };

    auto stage = [&](int s) {
        HiLo a0 = split8(xv0, xv1);
        HiLo a1 = split8(xv2, xv3);
        v8bf q0 = quant8(wv0, wv1, qPos, qNeg, qZero);
        v8bf q1 = quant8(wv2, wv3, qPos, qNeg, qZero);
        const int off = row * LDS_STRIDE + col;
        *(v8bf*)&sAhi[s][off]     = a0.hi;
        *(v8bf*)&sAhi[s][off + 8] = a1.hi;
        *(v8bf*)&sAlo[s][off]     = a0.lo;
        *(v8bf*)&sAlo[s][off + 8] = a1.lo;
        *(v8bf*)&sB  [s][off]     = q0;
        *(v8bf*)&sB  [s][off + 8] = q1;
    };

    v8f acc[4][2];
    const v8f vzero = {0.f, 0.f, 0.f, 0.f, 0.f, 0.f, 0.f, 0.f};
#pragma unroll
    for (int i = 0; i < 4; ++i)
#pragma unroll
        for (int j = 0; j < 2; ++j) acc[i][j] = vzero;

    // Per-lane fragment gather offsets (contiguous k = 16h + s, same order for A and B):
    // lanes 0-15 hold K=0..15 of row (lane), lanes 16-31 hold K=16..31.
    const int kfrag = (lane >> 4) * 16;
    const int frow  = lane & 15;

    auto ldfrag = [&](const bf16_t* base) -> v16bf {
        v8bf a = *(const v8bf*)(base);
        v8bf b = *(const v8bf*)(base + 8);
        return __builtin_shufflevector(a, b, 0, 1, 2, 3, 4, 5, 6, 7,
                                             8, 9, 10, 11, 12, 13, 14, 15);
    };

    auto compute = [&](int s) {
        v16bf bfr[2];
#pragma unroll
        for (int j = 0; j < 2; ++j)
            bfr[j] = ldfrag(&sB[s][(waveN + j * 16 + frow) * LDS_STRIDE + kfrag]);
#pragma unroll
        for (int i = 0; i < 4; ++i) {
            const int aoff = (waveM + i * 16 + frow) * LDS_STRIDE + kfrag;
            v16bf ahi = ldfrag(&sAhi[s][aoff]);
            v16bf alo = ldfrag(&sAlo[s][aoff]);
#pragma unroll
            for (int j = 0; j < 2; ++j) {
                acc[i][j] = __builtin_amdgcn_wmma_f32_16x16x32_bf16(
                    false, ahi, false, bfr[j], (short)0, acc[i][j], false, false);
                acc[i][j] = __builtin_amdgcn_wmma_f32_16x16x32_bf16(
                    false, alo, false, bfr[j], (short)0, acc[i][j], false, false);
            }
        }
    };

    const int NK = K_DIM / BK;               // 128 k-steps
    fetch(0);
    stage(0);
    __syncthreads();
#pragma unroll 1
    for (int kt = 0; kt < NK; ++kt) {
        const int cur = kt & 1;
        if (kt + 1 < NK) fetch(kt + 1);                     // global -> VGPR prefetch
        if (kt + 2 < NK) {                                  // L2 prefetch 2 steps ahead
            __builtin_prefetch(xg + (kt + 2) * BK, 0, 1);
            __builtin_prefetch(wg + (kt + 2) * BK, 0, 1);
        }
        compute(cur);                                       // ds gathers + 16 WMMAs
        if (kt + 1 < NK) {
            stage(cur ^ 1);                                 // write other buffer
            __syncthreads();                                // one barrier per k-step
        }
    }

    // Epilogue: C/D layout — VGPR r: lanes 0-15 -> M=r, lanes 16-31 -> M=r+8.
    const int cRow = (lane >> 4) * 8;
    const int cCol = lane & 15;
#pragma unroll
    for (int i = 0; i < 4; ++i)
#pragma unroll
        for (int j = 0; j < 2; ++j) {
            float* outp = Y + (size_t)(m0 + waveM + i * 16 + cRow) * N_DIM
                            + (n0 + waveN + j * 16 + cCol);
#pragma unroll
            for (int r = 0; r < 8; ++r)
                outp[(size_t)r * N_DIM] = acc[i][j][r];
        }
}

// ---------------------------------------------------------------------------
extern "C" void kernel_launch(void* const* d_in, const int* in_sizes, int n_in,
                              void* d_out, int out_size, void* d_ws, size_t ws_size,
                              hipStream_t stream) {
    const float* X = (const float*)d_in[0];      // (4*2048, 4096) f32
    const float* W = (const float*)d_in[1];      // (4096, 4096) f32
    float* Y = (float*)d_out;                    // (8192, 4096) f32

    float* partial = (float*)d_ws;               // RED_BLOCKS floats
    float* factor  = partial + RED_BLOCKS;       // 1 float

    const int wElems = in_sizes[1];              // 4096*4096
    const int M      = in_sizes[0] / K_DIM;      // 8192

    absmean_partial<<<RED_BLOCKS, NTHREADS, 0, stream>>>(W, partial, wElems / 4);
    finalize_factor<<<1, NTHREADS, 0, stream>>>(partial, factor, RED_BLOCKS,
                                                1.0f / (float)wElems);

    dim3 grid(N_DIM / BN, M / BM);               // (32, 64) workgroups
    ternary_gemm<<<grid, NTHREADS, 0, stream>>>(X, W, factor, Y);
}
